// DecoderClosedBook_5454608466756
// MI455X (gfx1250) — matrix operand order repaired and checked by reading, not compiled, not run on hardware
//
#include <hip/hip_runtime.h>
#include <hip/hip_bf16.h>

// ---------------------------------------------------------------------------
// DecoderClosedBook: embedding gather -> LSTM(512->1024) over T=128 -> 32000-way
// projection.  GEMMs via v_wmma_f32_16x16x32_f16; projection A tiles staged
// through LDS with global_load_async_to_lds_b128 (double buffered, ASYNCcnt).
// ---------------------------------------------------------------------------

#define VOCAB 32000
#define HID   512
#define LSTMH 1024
#define G4    4096          // 4 * LSTMH
#define BATCH 16
#define TSEQ  128
#define BT    2048          // BATCH * TSEQ

typedef __attribute__((ext_vector_type(16))) _Float16 v16h;
typedef __attribute__((ext_vector_type(8)))  _Float16 v8h;
typedef __attribute__((ext_vector_type(4)))  _Float16 v4h;
typedef __attribute__((ext_vector_type(8)))  float    v8f;

union AFrag { v16h v; v8h h[2]; };

// A-matrix (16x32 f16) fragment from row-major memory: lane m<16 rows M=m,
// K = {k0..k0+7, k0+16..k0+23}; lanes 16..31 K offset +8 (ISA 7.12.2).
__device__ __forceinline__ v16h load_a_frag(const _Float16* base, int ld, int k0) {
    const int lane = threadIdx.x & 31;
    const int row  = lane & 15;
    const int hi   = lane >> 4;
    const _Float16* p = base + row * ld + k0 + hi * 8;
    AFrag f;
    f.h[0] = *(const v8h*)(p);
    f.h[1] = *(const v8h*)(p + 16);
    return f.v;
}

// B-matrix (32x16 f16) fragment from row-major W[n][k]: lane n<16 -> col n,
// K = {k0..k0+7, k0+16..k0+23}; lanes 16..31 K offset +8.  Two 16-byte loads.
__device__ __forceinline__ v16h load_b_frag(const _Float16* base, int ld, int k0) {
    const int lane = threadIdx.x & 31;
    const int col  = lane & 15;
    const int hi   = lane >> 4;
    const _Float16* p = base + col * ld + k0 + hi * 8;
    AFrag f;
    f.h[0] = *(const v8h*)(p);
    f.h[1] = *(const v8h*)(p + 16);
    return f.v;
}

__device__ __forceinline__ float sigmoidf_(float x) {
    return 1.0f / (1.0f + __expf(-x));
}

// ---------------------------------------------------------------------------
// Workspace layout (bytes, 256-aligned)
// ---------------------------------------------------------------------------
#define WS_WIH   ((size_t)0)                          // 4096*512*2   =  4,194,304
#define WS_WHH   ((size_t)4194304)                    // 4096*1024*2  =  8,388,608
#define WS_OUTW  ((size_t)12582912)                   // 32000*1024*2 = 65,536,000
#define WS_EMB   ((size_t)78118912)                   // 2048*512*2   =  2,097,152
#define WS_GX    ((size_t)80216064)                   // 2048*4096*4  = 33,554,432
#define WS_HSEQ  ((size_t)113770496)                  // 129*16*1024*2=  4,227,072
#define WS_CNT   ((size_t)117997568)                  // 4

// ---------------------------------------------------------------------------
// Utility kernels
// ---------------------------------------------------------------------------
__global__ void zero_counter(unsigned* c) { *c = 0u; }

__global__ void f32_to_f16_v4(const float* __restrict__ src,
                              _Float16* __restrict__ dst, int n4) {
    int i = blockIdx.x * blockDim.x + threadIdx.x;
    if (i >= n4) return;
    const float4 v = ((const float4*)src)[i];
    v4h o = { (_Float16)v.x, (_Float16)v.y, (_Float16)v.z, (_Float16)v.w };
    ((v4h*)dst)[i] = o;
}

// Embedding gather with teacher forcing: row id = t*16 + b
__global__ void embed_kernel(const int* __restrict__ target,
                             const float* __restrict__ embedding,
                             _Float16* __restrict__ emb) {
    const int rowid = blockIdx.x;            // 0..2047
    const int t = rowid >> 4, b = rowid & 15;
    int idx = (t == 0) ? 1 : target[b * TSEQ + (t - 1)];   // BOS=1
    if (idx >= VOCAB || idx < 0) idx = 0;                  // UNK=0
    const float* src = embedding + (size_t)idx * HID;
    _Float16* dst = emb + (size_t)rowid * HID;
    for (int k = threadIdx.x; k < HID; k += blockDim.x)
        dst[k] = (_Float16)src[k];
}

// ---------------------------------------------------------------------------
// GEMM 1: gates_x[t*16+b][g] = emb @ w_ih^T + (b_ih + b_hh)
// grid (128 Mtiles, 64), block 128 (4 waves), wave -> one 16x16 N tile, K=512
// ---------------------------------------------------------------------------
__global__ __launch_bounds__(128)
void gemm_gates_x(const _Float16* __restrict__ emb,      // [2048][512]
                  const _Float16* __restrict__ wih,      // [4096][512]
                  const float* __restrict__ b_ih,
                  const float* __restrict__ b_hh,
                  float* __restrict__ gates_x) {         // [2048][4096]
    const int wave = threadIdx.x >> 5;
    const int lane = threadIdx.x & 31;
    const int mt = blockIdx.x;                  // 0..127
    const int nt = blockIdx.y * 4 + wave;       // 0..255
    const _Float16* A  = emb + (size_t)mt * 16 * HID;
    const _Float16* Bp = wih + (size_t)nt * 16 * HID;
    v8f acc = {};
    for (int k0 = 0; k0 < HID; k0 += 32) {
        v16h a = load_a_frag(A, HID, k0);
        v16h b = load_b_frag(Bp, HID, k0);
        acc = __builtin_amdgcn_wmma_f32_16x16x32_f16(false, a, false, b,
                                                     (short)0, acc, false, false);
    }
    const int col = nt * 16 + (lane & 15);
    const float bias = b_ih[col] + b_hh[col];
    const int hi = lane >> 4;
    float* outp = gates_x + col;
#pragma unroll
    for (int r = 0; r < 8; ++r) {
        const int m = mt * 16 + r + hi * 8;
        outp[(size_t)m * G4] = acc[r] + bias;
    }
}

// ---------------------------------------------------------------------------
// Persistent LSTM: 64 workgroups x 128 threads.  Workgroup wg owns hidden
// columns [wg*16, wg*16+16); wave w computes gate block w (i/f/g/o).
// Cell state c lives in registers for all 128 steps.  Grid barrier per step.
// ---------------------------------------------------------------------------
__global__ __launch_bounds__(128)
void lstm_persistent(const _Float16* __restrict__ whh,     // [4096][1024] f16
                     const float* __restrict__ gates_x,    // [2048][4096]
                     const float* __restrict__ c0,         // [16][1024]
                     _Float16* __restrict__ h_seq,         // [(T+1)*16][1024] f16
                     unsigned* __restrict__ counter) {
    __shared__ float lds[4 * 256];
    const int tid  = threadIdx.x;
    const int wave = tid >> 5;
    const int lane = tid & 31;
    const int wg   = blockIdx.x;                 // 0..63
    const int col  = lane & 15;
    const int hi   = lane >> 4;
    const int gcol = wave * LSTMH + wg * 16 + col;

    // two (batch, hidden-j) pairs per thread hold persistent cell state
    const int b0 = tid >> 4,          j0 = tid & 15;
    const int b1 = (tid + 128) >> 4,  j1 = tid & 15;
    float c_0 = c0[b0 * LSTMH + wg * 16 + j0];
    float c_1 = c0[b1 * LSTMH + wg * 16 + j1];

    const _Float16* Bp = whh + (size_t)(wave * LSTMH + wg * 16) * LSTMH;

    for (int t = 0; t < TSEQ; ++t) {
        const _Float16* A = h_seq + (size_t)t * 16 * LSTMH;   // h at step t (16x1024)
        v8f acc = {};
#pragma unroll 4
        for (int k0 = 0; k0 < LSTMH; k0 += 32) {
            v16h a = load_a_frag(A, LSTMH, k0);
            v16h b = load_b_frag(Bp, LSTMH, k0);
            acc = __builtin_amdgcn_wmma_f32_16x16x32_f16(false, a, false, b,
                                                         (short)0, acc, false, false);
        }
        // add precomputed x-gates (+bias), park gate tile in LDS [gate][b][j]
        const float* gx = gates_x + (size_t)t * 16 * G4 + gcol;
#pragma unroll
        for (int r = 0; r < 8; ++r) {
            const int mb = r + hi * 8;                       // batch row
            lds[wave * 256 + mb * 16 + col] = acc[r] + gx[(size_t)mb * G4];
        }
        __syncthreads();
        // pointwise: i,f,g,o -> c,h   (PyTorch gate order)
        {
            float ig = sigmoidf_(lds[0 * 256 + b0 * 16 + j0]);
            float fg = sigmoidf_(lds[1 * 256 + b0 * 16 + j0]);
            float gg = tanhf   (lds[2 * 256 + b0 * 16 + j0]);
            float og = sigmoidf_(lds[3 * 256 + b0 * 16 + j0]);
            c_0 = fg * c_0 + ig * gg;
            float h = og * tanhf(c_0);
            h_seq[(size_t)((t + 1) * 16 + b0) * LSTMH + wg * 16 + j0] = (_Float16)h;
        }
        {
            float ig = sigmoidf_(lds[0 * 256 + b1 * 16 + j1]);
            float fg = sigmoidf_(lds[1 * 256 + b1 * 16 + j1]);
            float gg = tanhf   (lds[2 * 256 + b1 * 16 + j1]);
            float og = sigmoidf_(lds[3 * 256 + b1 * 16 + j1]);
            c_1 = fg * c_1 + ig * gg;
            float h = og * tanhf(c_1);
            h_seq[(size_t)((t + 1) * 16 + b1) * LSTMH + wg * 16 + j1] = (_Float16)h;
        }
        // device-wide barrier (monotonic counter; re-zeroed each launch)
        __threadfence();
        __syncthreads();
        if (tid == 0) {
            atomicAdd(counter, 1u);
            const unsigned tgt = 64u * (unsigned)(t + 1);
            while (__hip_atomic_load(counter, __ATOMIC_RELAXED,
                                     __HIP_MEMORY_SCOPE_AGENT) < tgt)
                __builtin_amdgcn_s_sleep(8);
        }
        __syncthreads();
        __threadfence();
    }
}

// ---------------------------------------------------------------------------
// GEMM 2: logits[b*T+t][v] = hs @ out_w^T + out_b
// grid (500 Nblocks, 4 Mquads), block 128 (4 waves; wave -> 16 cols).
// A (hs) tiles for 4 M-tiles are staged cooperatively into LDS with
// global_load_async_to_lds_b128 (double buffered, K-chunks of 128), so the 4
// waves share one copy; B (out_w) streams straight from global into WMMA.
// ---------------------------------------------------------------------------
#define KCP      128                 // K chunk (halfs)
#define ASTRIDE  136                 // padded LDS row stride in halfs (272 B)
#define ABUF_H   (64 * ASTRIDE)      // halfs per buffer (17,408 B)

__device__ __forceinline__ v16h load_a_lds(const _Float16* Ab, int rowbase,
                                           int kl, int lane, int hi) {
    const _Float16* p = Ab + (rowbase + (lane & 15)) * ASTRIDE + kl + hi * 8;
    AFrag f;
    f.h[0] = *(const v8h*)(p);
    f.h[1] = *(const v8h*)(p + 16);
    return f.v;
}

__global__ __launch_bounds__(128)
void gemm_logits(const _Float16* __restrict__ hseq,   // [(T+1)*16][1024]
                 const _Float16* __restrict__ outw,   // [32000][1024]
                 const float* __restrict__ outb,
                 float* __restrict__ outp) {          // [2048][32000]
    __shared__ __align__(16) _Float16 Alds[2 * ABUF_H];   // 34,816 bytes
    const int tid  = threadIdx.x;
    const int wave = tid >> 5;
    const int lane = tid & 31;
    const int nt = blockIdx.x * 4 + wave;             // 0..1999
    const int mquad = blockIdx.y;                     // 0..3
    const _Float16* Bp = outw + (size_t)nt * 16 * LSTMH;
    const int col = nt * 16 + (lane & 15);
    const float bias = outb[col];
    const int hi = lane >> 4;
    const unsigned ldsbase = (unsigned)(uintptr_t)Alds;   // low 32 bits = LDS offset

    for (int mb = 0; mb < 8; ++mb) {
        const int mt0 = mquad * 32 + mb * 4;          // base M tile (4 tiles/block)
        const int mrow0 = mt0 * 16;                   // 64 consecutive output rows
        v8f acc0 = {}, acc1 = {}, acc2 = {}, acc3 = {};

        // Stage A[64 rows][KCP halfs] for chunk kc into buffer buf.
        // Each wave issues 8 async b128 ops (ASYNCcnt += 8 per wave).
        auto stage = [&](int kc, int buf) {
#pragma unroll
            for (int i = 0; i < 8; ++i) {
                const int li   = tid + i * 128;       // 0..1023
                const int rrow = li >> 4;             // 0..63
                const int seg  = li & 15;             // 16B segment in row
                const int m  = mrow0 + rrow;          // output row = b*T + t
                const int tt = m & (TSEQ - 1);
                const int bb = m >> 7;
                const _Float16* g = hseq +
                    (size_t)((tt + 1) * 16 + bb) * LSTMH + kc + seg * 8;
                const unsigned la = ldsbase +
                    (unsigned)(buf * ABUF_H + rrow * ASTRIDE + seg * 8) * 2u;
                asm volatile("global_load_async_to_lds_b128 %0, %1, off"
                             :: "v"(la), "v"((unsigned long long)(uintptr_t)g)
                             : "memory");
            }
        };

        stage(0, 0);
        for (int c = 0; c < 8; ++c) {
            if (c < 7) {
                stage((c + 1) * KCP, (c + 1) & 1);    // prefetch next chunk
                asm volatile("s_wait_asynccnt 0x8" ::: "memory");  // oldest 8 done
            } else {
                asm volatile("s_wait_asynccnt 0x0" ::: "memory");
            }
            __syncthreads();
            const _Float16* Ab = Alds + (c & 1) * ABUF_H;
#pragma unroll
            for (int kl = 0; kl < KCP; kl += 32) {
                v16h bf = load_b_frag(Bp, LSTMH, c * KCP + kl);
                v16h a0 = load_a_lds(Ab, 0,  kl, lane, hi);
                v16h a1 = load_a_lds(Ab, 16, kl, lane, hi);
                v16h a2 = load_a_lds(Ab, 32, kl, lane, hi);
                v16h a3 = load_a_lds(Ab, 48, kl, lane, hi);
                acc0 = __builtin_amdgcn_wmma_f32_16x16x32_f16(false, a0, false, bf,
                                                              (short)0, acc0, false, false);
                acc1 = __builtin_amdgcn_wmma_f32_16x16x32_f16(false, a1, false, bf,
                                                              (short)0, acc1, false, false);
                acc2 = __builtin_amdgcn_wmma_f32_16x16x32_f16(false, a2, false, bf,
                                                              (short)0, acc2, false, false);
                acc3 = __builtin_amdgcn_wmma_f32_16x16x32_f16(false, a3, false, bf,
                                                              (short)0, acc3, false, false);
            }
            __syncthreads();   // buffer may be overwritten next iteration
        }

        v8f accs[4] = { acc0, acc1, acc2, acc3 };
#pragma unroll
        for (int i = 0; i < 4; ++i) {
            const int mbase = (mt0 + i) * 16;
#pragma unroll
            for (int r = 0; r < 8; ++r) {
                const int m = mbase + r + hi * 8;
                outp[(size_t)m * VOCAB + col] = accs[i][r] + bias;
            }
        }
    }
}

// ---------------------------------------------------------------------------
extern "C" void kernel_launch(void* const* d_in, const int* in_sizes, int n_in,
                              void* d_out, int out_size, void* d_ws, size_t ws_size,
                              hipStream_t stream) {
    const float* h0        = (const float*)d_in[0];
    const float* c0        = (const float*)d_in[1];
    const int*   target    = (const int*)  d_in[2];
    const float* embedding = (const float*)d_in[3];
    const float* w_ih      = (const float*)d_in[4];
    const float* w_hh      = (const float*)d_in[5];
    const float* b_ih      = (const float*)d_in[6];
    const float* b_hh      = (const float*)d_in[7];
    const float* out_w     = (const float*)d_in[8];
    const float* out_b     = (const float*)d_in[9];
    float* out = (float*)d_out;

    char* ws = (char*)d_ws;
    _Float16* wih16  = (_Float16*)(ws + WS_WIH);
    _Float16* whh16  = (_Float16*)(ws + WS_WHH);
    _Float16* outw16 = (_Float16*)(ws + WS_OUTW);
    _Float16* emb16  = (_Float16*)(ws + WS_EMB);
    float*    gx     = (float*)   (ws + WS_GX);
    _Float16* hseq   = (_Float16*)(ws + WS_HSEQ);
    unsigned* cnt    = (unsigned*)(ws + WS_CNT);

    zero_counter<<<1, 1, 0, stream>>>(cnt);

    // f32 -> f16 weight/state conversions
    {
        int n4 = G4 * HID / 4;       // w_ih
        f32_to_f16_v4<<<(n4 + 255) / 256, 256, 0, stream>>>(w_ih, wih16, n4);
    }
    {
        int n4 = G4 * LSTMH / 4;     // w_hh
        f32_to_f16_v4<<<(n4 + 255) / 256, 256, 0, stream>>>(w_hh, whh16, n4);
    }
    {
        int n4 = VOCAB * LSTMH / 4;  // out_w
        f32_to_f16_v4<<<(n4 + 255) / 256, 256, 0, stream>>>(out_w, outw16, n4);
    }
    {
        int n4 = BATCH * LSTMH / 4;  // h0 -> h_seq[0]
        f32_to_f16_v4<<<(n4 + 255) / 256, 256, 0, stream>>>(h0, hseq, n4);
    }

    embed_kernel<<<BT, 128, 0, stream>>>(target, embedding, emb16);

    gemm_gates_x<<<dim3(128, 64), 128, 0, stream>>>(emb16, wih16, b_ih, b_hh, gx);

    lstm_persistent<<<64, 128, 0, stream>>>(whh16, gx, c0, hseq, cnt);

    gemm_logits<<<dim3(500, 4), 128, 0, stream>>>(hseq, outw16, out_b, out);
}
